// TMUModel_3246995275914
// MI455X (gfx1250) — matrix operand order repaired and checked by reading, not compiled, not run on hardware
//
#include <hip/hip_runtime.h>
#include <hip/hip_bf16.h>

typedef __bf16 bf16t;
typedef __attribute__((ext_vector_type(16))) __bf16 v16bf;
typedef __attribute__((ext_vector_type(8)))  __bf16 v8bf;
typedef __attribute__((ext_vector_type(8)))  float  v8f;
typedef __attribute__((ext_vector_type(4)))  float  v4f;

#define Bsz   64
#define Tsz   1024
#define Hsz   512
#define NWG   64        // persistent recurrent grid: 32 H-slices x 2 LSTMs
#define NTHR  128       // 4 waves of 32
#define NWAVE 4

// workspace layout (bytes)
#define WS_BAR   0                         // unsigned barrier counter
#define WS_PART  256                       // 64 float loss partials
#define WS_H     4096                      // 2 parity x 2 lstm x 64 x 512 bf16 (256 KB)
#define WS_XBF   (512 * 1024)              // x1,x2 converted to bf16: 2 x 64x1024x512 (128 MB)
#define HBUF_ELEMS (2 * Bsz * Hsz)         // per parity (both LSTMs)
#define XELEMS  (Bsz * Tsz * Hsz)          // per LSTM

static __device__ __forceinline__ float bf2f(bf16t b) {
  unsigned short s = __builtin_bit_cast(unsigned short, b);
  unsigned u = ((unsigned)s) << 16;
  return __builtin_bit_cast(float, u);
}
static __device__ __forceinline__ bf16t f2bf(float f) {
  unsigned u = __builtin_bit_cast(unsigned, f);
  unsigned r = (u + 0x7FFFu + ((u >> 16) & 1u)) >> 16;
  unsigned short s = (unsigned short)r;
  return __builtin_bit_cast(bf16t, s);
}
static __device__ __forceinline__ float sigmoid_fast(float x) {
  return 1.0f / (1.0f + __expf(-x));
}
static __device__ __forceinline__ float tanh_fast(float x) {
  return 2.0f / (1.0f + __expf(-2.0f * x)) - 1.0f;
}

// device-scope monotonic counter barrier (reset by ws_init each launch)
static __device__ __forceinline__ void grid_barrier(unsigned* bar, unsigned target) {
  __threadfence();
  __syncthreads();
  if (threadIdx.x == 0) {
    __hip_atomic_fetch_add(bar, 1u, __ATOMIC_RELEASE, __HIP_MEMORY_SCOPE_AGENT);
    while (__hip_atomic_load(bar, __ATOMIC_ACQUIRE, __HIP_MEMORY_SCOPE_AGENT) < target) {
      __builtin_amdgcn_s_sleep(2);
    }
  }
  __syncthreads();
}

__global__ void ws_init(unsigned char* ws) {
  if (threadIdx.x == 0) *(unsigned*)(ws + WS_BAR) = 0u;
}

// Fully parallel, bandwidth-bound fp32 -> bf16 conversion of x1 and x2.
// Runs on the whole chip once; removes all conversion VALU from the serial loop.
__global__ void x_to_bf16(const float* __restrict__ x1,
                          const float* __restrict__ x2,
                          unsigned char* ws)
{
  bf16t* xbf = (bf16t*)(ws + WS_XBF);
  const size_t n4 = (size_t)XELEMS / 4;            // vec4 chunks per LSTM
  size_t stride = (size_t)gridDim.x * blockDim.x;
  for (size_t i = (size_t)blockIdx.x * blockDim.x + threadIdx.x; i < n4; i += stride) {
    v4f a = *(const v4f*)(x1 + i * 4);
    v4f b = *(const v4f*)(x2 + i * 4);
    union { bf16t e[4]; unsigned long long u; } pa, pb;
#pragma unroll
    for (int j = 0; j < 4; ++j) { pa.e[j] = f2bf(a[j]); pb.e[j] = f2bf(b[j]); }
    *(unsigned long long*)(xbf + i * 4)           = pa.u;
    *(unsigned long long*)(xbf + XELEMS + i * 4)  = pb.u;
  }
}

__global__ void lstm_recurrent(const float* __restrict__ mem,
                               const float* __restrict__ wx1, const float* __restrict__ bx1,
                               const float* __restrict__ wh1, const float* __restrict__ bh1,
                               const float* __restrict__ wx2, const float* __restrict__ bx2,
                               const float* __restrict__ wh2, const float* __restrict__ bh2,
                               unsigned char* ws)
{
  extern __shared__ char smem[];
  bf16t* wx_lds = (bf16t*)smem;                         // 64 gate-cols x 512 K (bf16)
  bf16t* wh_lds = (bf16t*)(smem + 64 * Hsz * 2);        // 64 x 512 (bf16)
  float* red    = (float*)(smem + 2 * 64 * Hsz * 2);    // 128 floats

  unsigned* bar  = (unsigned*)(ws + WS_BAR);
  float*    part = (float*)(ws + WS_PART);
  bf16t*    hbuf = (bf16t*)(ws + WS_H);
  const bf16t* xbf = (const bf16t*)(ws + WS_XBF);

  const int wg      = blockIdx.x;
  const int lstm    = wg >> 5;               // 0 or 1
  const int colbase = (wg & 31) * 16;        // this WG's 16-wide H-slice
  const int tid  = threadIdx.x;
  const int wave = tid >> 5;
  const int lane = tid & 31;

  const float* wx = lstm ? wx2 : wx1;
  const float* wh = lstm ? wh2 : wh1;
  const float* bx = lstm ? bx2 : bx1;
  const float* bh = lstm ? bh2 : bh1;
  const bf16t* x  = xbf + (size_t)lstm * XELEMS;

  // Stage weight slices into LDS as bf16: local row lr = gate*16 + c maps to
  // global gate-column j = gate*512 + colbase + c. One row = 512 contiguous K.
  for (int idx = tid; idx < 64 * Hsz; idx += NTHR) {
    int lr = idx >> 9;
    int k  = idx & (Hsz - 1);
    int j  = (lr >> 4) * Hsz + colbase + (lr & 15);
    wx_lds[idx] = f2bf(wx[(size_t)j * Hsz + k]);
    wh_lds[idx] = f2bf(wh[(size_t)j * Hsz + k]);
  }

  // per-lane fused bias (bx + bh) for the 4 gates at this lane's output column
  const int n = lane & 15;
  float bias[4];
#pragma unroll
  for (int g = 0; g < 4; ++g) {
    int j = g * Hsz + colbase + n;
    bias[g] = bx[j] + bh[j];
  }

  // zero-init h (parity 0) for this wave's 16x16 patch
  const int m0 = wave * 16;
  {
    bf16t z = f2bf(0.0f);
#pragma unroll
    for (int e = 0; e < 8; ++e) {
      int m = m0 + e + ((lane >> 4) << 3);
      hbuf[(size_t)lstm * Bsz * Hsz + (size_t)m * Hsz + colbase + n] = z;
    }
  }
  __syncthreads();
  grid_barrier(bar, NWG);   // barrier #0: h0 + LDS weights ready everywhere

  const int mrow  = m0 + (lane & 15);      // A-matrix row this lane feeds
  const int koffA = ((lane >> 4) << 3);    // 0 / 8   (16-bit A 16x32 layout)
  const int koffB = ((lane >> 4) << 4);    // 0 / 16  (unused; B loads use koffA)

  float c_state[8];
#pragma unroll
  for (int e = 0; e < 8; ++e) c_state[e] = 0.0f;
  float lacc = 0.0f;

  for (int t = 0; t < Tsz; ++t) {
    const int p = t & 1, q = p ^ 1;
    v8f acc[4];
#pragma unroll
    for (int g = 0; g < 4; ++g) { v8f z = {}; acc[g] = z; }

    // ---- x-half of the gates: A = xbf[:, t, :] (pure 16B bf16 loads) ----
    const bf16t* xrow = x + (size_t)mrow * (Tsz * Hsz) + (size_t)t * Hsz;
    for (int kt = 0; kt < 16; ++kt) {
      int klo = kt * 32 + koffA;
      union { v16bf v; v8bf h[2]; } a;
      a.h[0] = *(const v8bf*)(xrow + klo);
      a.h[1] = *(const v8bf*)(xrow + klo + 16);
      int kb = kt * 32 + koffA;
      // preload all 4 gate B-fragments (8 clause-able ds_load_b128) ...
      union { v16bf v; v8bf h[2]; } b[4];
#pragma unroll
      for (int g = 0; g < 4; ++g) {
        const bf16t* wp = wx_lds + ((g * 16 + n) * Hsz + kb);
        b[g].h[0] = *(const v8bf*)wp;
        b[g].h[1] = *(const v8bf*)(wp + 16);
      }
      // ... then drain into 4 back-to-back WMMAs
#pragma unroll
      for (int g = 0; g < 4; ++g) {
        acc[g] = __builtin_amdgcn_wmma_f32_16x16x32_bf16(
            false, a.v, false, b[g].v, (short)0, acc[g], false, false);
      }
    }

    // ---- h-half of the gates: A = h_{t-1} (bf16 in L2-resident buffer) ----
    const bf16t* hrow = hbuf + (size_t)p * HBUF_ELEMS +
                        (size_t)lstm * Bsz * Hsz + (size_t)mrow * Hsz;
    for (int kt = 0; kt < 16; ++kt) {
      int klo = kt * 32 + koffA;
      union { v16bf v; v8bf h[2]; } a;
      a.h[0] = *(const v8bf*)(hrow + klo);
      a.h[1] = *(const v8bf*)(hrow + klo + 16);
      int kb = kt * 32 + koffA;
      union { v16bf v; v8bf h[2]; } b[4];
#pragma unroll
      for (int g = 0; g < 4; ++g) {
        const bf16t* wp = wh_lds + ((g * 16 + n) * Hsz + kb);
        b[g].h[0] = *(const v8bf*)wp;
        b[g].h[1] = *(const v8bf*)(wp + 16);
      }
#pragma unroll
      for (int g = 0; g < 4; ++g) {
        acc[g] = __builtin_amdgcn_wmma_f32_16x16x32_bf16(
            false, a.v, false, b[g].v, (short)0, acc[g], false, false);
      }
    }

    if (t + 1 < Tsz) __builtin_prefetch(xrow + Hsz, 0, 1);  // next step's x row

    // ---- elementwise LSTM update; c stays in registers all 1024 steps ----
    bf16t* hout = hbuf + (size_t)q * HBUF_ELEMS + (size_t)lstm * Bsz * Hsz;
#pragma unroll
    for (int e = 0; e < 8; ++e) {
      float iv = sigmoid_fast(acc[0][e] + bias[0]);
      float fv = sigmoid_fast(acc[1][e] + bias[1]);
      float gv = tanh_fast   (acc[2][e] + bias[2]);
      float ov = sigmoid_fast(acc[3][e] + bias[3]);
      float c  = c_state[e] * fv + iv * gv;
      c_state[e] = c;
      float h  = ov * tanh_fast(c);
      int m = m0 + e + ((lane >> 4) << 3);
      hout[(size_t)m * Hsz + colbase + n] = f2bf(h);
      if (lstm == 1) {
        float mv = mem[(size_t)m * (Tsz * Hsz) + (size_t)t * Hsz + colbase + n];
        float d = c - mv;
        lacc += d * d;
      }
    }

    grid_barrier(bar, (unsigned)(NWG * (t + 2)));
  }

  // deterministic loss partial per workgroup (fixed-order LDS reduce)
  red[tid] = lacc;
  __syncthreads();
  if (tid == 0) {
    float s = 0.0f;
    for (int i = 0; i < NTHR; ++i) s += red[i];
    part[wg] = s;
  }
}

__global__ void lstm_final(const float* __restrict__ wfc, const float* __restrict__ bfc,
                           float* __restrict__ out, const unsigned char* __restrict__ ws)
{
  const float* part = (const float*)(ws + WS_PART);
  const bf16t* hbuf = (const bf16t*)(ws + WS_H);
  const int pf = ((Tsz & 1) ? 1 : 0);                 // last write parity (0 for even T)
  const bf16t* h1 = hbuf + (size_t)pf * HBUF_ELEMS;
  const bf16t* h2 = h1 + Bsz * Hsz;

  const int tid = threadIdx.x, wave = tid >> 5, lane = tid & 31;
  const int tile = blockIdx.x * NWAVE + wave;         // 128 tiles: 4 rows x 32 cols
  const int rt = tile & 3, ct = tile >> 2;
  const int m0 = rt * 16, n0 = ct * 16;
  const int n = lane & 15;
  const int arow  = m0 + (lane & 15);
  const int koffA = ((lane >> 4) << 3);
  const int koffB = ((lane >> 4) << 4);

  v8f acc = {};
  const bf16t* h1r = h1 + (size_t)arow * Hsz;
  const bf16t* h2r = h2 + (size_t)arow * Hsz;
  const float* wrow = wfc + (size_t)(n0 + n) * Hsz;

  for (int kt = 0; kt < 16; ++kt) {
    int klo = kt * 32 + koffA;
    union { v16bf v; bf16t e[16]; } a;
#pragma unroll
    for (int i = 0; i < 8; ++i) {
      a.e[i]     = f2bf(bf2f(h1r[klo + i])      + bf2f(h2r[klo + i]));
      a.e[8 + i] = f2bf(bf2f(h1r[klo + 16 + i]) + bf2f(h2r[klo + 16 + i]));
    }
    int kb = kt * 32 + koffA;
    union { v16bf v; bf16t e[16]; } b;
#pragma unroll
    for (int i = 0; i < 8; ++i) { b.e[i] = f2bf(wrow[kb + i]); b.e[8 + i] = f2bf(wrow[kb + 16 + i]); }
    acc = __builtin_amdgcn_wmma_f32_16x16x32_bf16(
        false, a.v, false, b.v, (short)0, acc, false, false);
  }

  float bias = bfc[n0 + n];
#pragma unroll
  for (int e = 0; e < 8; ++e) {
    int m = m0 + e + ((lane >> 4) << 3);
    out[(size_t)m * Hsz + n0 + n] = acc[e] + bias;
  }

  // fixed-order loss reduction -> scalar at out[B*O]
  if (blockIdx.x == 0 && tid == 0) {
    float s = 0.0f;
    for (int i = 0; i < NWG; ++i) s += part[i];
    out[Bsz * Hsz] = s / (float)(Bsz * Hsz);
  }
}

extern "C" void kernel_launch(void* const* d_in, const int* in_sizes, int n_in,
                              void* d_out, int out_size, void* d_ws, size_t ws_size,
                              hipStream_t stream) {
  const float* x1   = (const float*)d_in[0];
  const float* x2   = (const float*)d_in[1];
  const float* mem  = (const float*)d_in[2];
  const float* wx1  = (const float*)d_in[3];
  const float* bx1v = (const float*)d_in[4];
  const float* wh1  = (const float*)d_in[5];
  const float* bh1v = (const float*)d_in[6];
  const float* wx2  = (const float*)d_in[7];
  const float* bx2v = (const float*)d_in[8];
  const float* wh2  = (const float*)d_in[9];
  const float* bh2v = (const float*)d_in[10];
  const float* wfc  = (const float*)d_in[11];
  const float* bfc  = (const float*)d_in[12];
  float* out = (float*)d_out;
  unsigned char* ws = (unsigned char*)d_ws;

  hipLaunchKernelGGL(ws_init, dim3(1), dim3(64), 0, stream, ws);

  // whole-chip bandwidth-bound conversion: 268 MB read + 134 MB write (~20 us)
  hipLaunchKernelGGL(x_to_bf16, dim3(4096), dim3(256), 0, stream, x1, x2, ws);

  size_t smem = 2 * 64 * Hsz * 2 /*bf16 W slices*/ + NTHR * sizeof(float);
  hipLaunchKernelGGL(lstm_recurrent, dim3(NWG), dim3(NTHR), smem, stream,
                     mem, wx1, bx1v, wh1, bh1v,
                     wx2, bx2v, wh2, bh2v, ws);

  hipLaunchKernelGGL(lstm_final, dim3(32), dim3(NTHR), 0, stream, wfc, bfc, out, ws);
}
